// MoE_74045236183586
// MI455X (gfx1250) — matrix operand order repaired and checked by reading, not compiled, not run on hardware
//
#include <hip/hip_runtime.h>
#include <hip/hip_bf16.h>

// ---------------------------------------------------------------------------
// MoE (bs=8192, d=512, o=512, 32 experts, top-4), output = batch-mean [512].
// f32 routing -> per-expert token lists -> bf16 WMMA grouped GEMM with
// gate-weighted column reduction (mean folded into gates). Only selected
// 4/32 experts computed (8x fewer FLOPs than the reference's dense einsum).
// GEMM tile: M=64 x N=128, K stepped by 32; 8 waves, each wave = 16x64 of C
// (1 A-frag reused over 4 B-frags -> 4 WMMA per K-step per wave).
// Data movement: GLOBAL_LOAD_ASYNC_TO_LDS (ASYNCcnt, double-buffered) when
// the builtin exists, else register-staged stores (compiler-pipelined).
// ---------------------------------------------------------------------------

#define BS      8192
#define DIM     512
#define ODIM    512
#define NEXP    32
#define TOPK    4
#define CAP     8192          // worst-case tokens per expert (indices unique per token)
#define KSTEPS  (DIM / 32)    // 16

typedef __attribute__((ext_vector_type(16))) __bf16 v16bf;
typedef __attribute__((ext_vector_type(8)))  float  v8f;

// pointer types for the async-to-LDS builtin: b128 <-> GCC int4 vector
typedef int v4i_vs __attribute__((vector_size(16)));
typedef __attribute__((address_space(1))) v4i_vs* gvp_b128;   // global (AS1)
typedef __attribute__((address_space(3))) v4i_vs* lvp_b128;   // LDS    (AS3)

union FragU {                 // 32-byte bf16 WMMA A/B fragment
    uint4 u[2];
    v16bf v;
};

#if __has_builtin(__builtin_amdgcn_global_load_async_to_lds_b128)
#define HAVE_ASYNC_LDS 1
#else
#define HAVE_ASYNC_LDS 0
#endif

__device__ __forceinline__ void wait_async_le3() {
#if __has_builtin(__builtin_amdgcn_s_wait_asynccnt)
    __builtin_amdgcn_s_wait_asynccnt(3);
#else
    asm volatile("s_wait_asynccnt 0x3" ::: "memory");
#endif
}
__device__ __forceinline__ void wait_async_le0() {
#if __has_builtin(__builtin_amdgcn_s_wait_asynccnt)
    __builtin_amdgcn_s_wait_asynccnt(0);
#else
    asm volatile("s_wait_asynccnt 0x0" ::: "memory");
#endif
}

__device__ __forceinline__ unsigned short f32_to_bf16(float f) {
    unsigned int u = __builtin_bit_cast(unsigned int, f);
    u += 0x7fffu + ((u >> 16) & 1u);   // round-to-nearest-even
    return (unsigned short)(u >> 16);
}

// --- 1) zero output accumulator + per-expert counters ----------------------
__global__ void moe_init_kernel(float* __restrict__ out, int* __restrict__ cnt) {
    int t = threadIdx.x;
    if (t < ODIM) out[t] = 0.0f;
    if (t < NEXP) cnt[t] = 0;
}

// --- 2/3) f32 -> bf16 conversion (float4 -> ushort4) -----------------------
__global__ void cvt_bf16_kernel(const float* __restrict__ src,
                                unsigned short* __restrict__ dst, int n4) {
    int i = blockIdx.x * blockDim.x + threadIdx.x;
    if (i >= n4) return;
    float4 f = ((const float4*)src)[i];
    ushort4 o;
    o.x = f32_to_bf16(f.x);
    o.y = f32_to_bf16(f.y);
    o.z = f32_to_bf16(f.z);
    o.w = f32_to_bf16(f.w);
    ((ushort4*)dst)[i] = o;
}

// --- 4) routing: one wave32 per token --------------------------------------
__global__ __launch_bounds__(256) void moe_route_kernel(
    const float* __restrict__ x,   // [BS, DIM]
    const float* __restrict__ Wg,  // [NEXP, DIM]
    const float* __restrict__ Wn,  // [NEXP, DIM]
    int*   __restrict__ cnt,       // [NEXP]
    int*   __restrict__ tok_list,  // [NEXP, CAP]
    float* __restrict__ gate_list) // [NEXP, CAP]
{
    __shared__ float xs_all[8 * DIM];       // 16 KB: one row per wave
    const int wave = threadIdx.x >> 5;
    const int lane = threadIdx.x & 31;
    const int b    = blockIdx.x * 8 + wave;

    float* xs = &xs_all[wave * DIM];
    for (int i = lane; i < DIM; i += 32)
        xs[i] = x[(size_t)b * DIM + i];
    // wave-private LDS region: same-wave DS ordering suffices, no block barrier

    // lane == expert: dual dot products with float4 loads
    const float4* x4 = (const float4*)xs;
    const float4* g4 = (const float4*)(Wg + (size_t)lane * DIM);
    const float4* n4 = (const float4*)(Wn + (size_t)lane * DIM);
    float ag = 0.f, an = 0.f;
    #pragma unroll 4
    for (int i = 0; i < DIM / 4; ++i) {
        float4 xv = x4[i], gv = g4[i], nv = n4[i];
        ag += xv.x * gv.x + xv.y * gv.y + xv.z * gv.z + xv.w * gv.w;
        an += xv.x * nv.x + xv.y * nv.y + xv.z * nv.z + xv.w * nv.w;
    }
    float sp    = (an > 20.f) ? an : log1pf(__expf(an));   // softplus
    float logit = ag + sp;

    // softmax over the 32 experts (wave-wide)
    float m = logit;
    for (int o = 16; o >= 1; o >>= 1) m = fmaxf(m, __shfl_xor(m, o, 32));
    float p = __expf(logit - m);
    float s = p;
    for (int o = 16; o >= 1; o >>= 1) s += __shfl_xor(s, o, 32);
    float prob = p / s;

    // top-4 (descending) via 4 argmax butterflies
    float v = prob;
    float tv[TOPK]; int ti[TOPK];
    #pragma unroll
    for (int k = 0; k < TOPK; ++k) {
        float bv = v; int bi = lane;
        for (int o = 16; o >= 1; o >>= 1) {
            float ov = __shfl_xor(bv, o, 32);
            int   oi = __shfl_xor(bi, o, 32);
            if (ov > bv || (ov == bv && oi < bi)) { bv = ov; bi = oi; }
        }
        tv[k] = bv; ti[k] = bi;
        if (lane == bi) v = -1e30f;
    }
    // reference quirk: descending probs paired positionally with
    // ascending-sorted expert indices -> sort indices only.
    int s0 = ti[0], s1 = ti[1], s2 = ti[2], s3 = ti[3], tmp;
    if (s0 > s1) { tmp = s0; s0 = s1; s1 = tmp; }
    if (s2 > s3) { tmp = s2; s2 = s3; s3 = tmp; }
    if (s0 > s2) { tmp = s0; s0 = s2; s2 = tmp; }
    if (s1 > s3) { tmp = s1; s1 = s3; s3 = tmp; }
    if (s1 > s2) { tmp = s1; s1 = s2; s2 = tmp; }

    if (lane == 0) {
        const float inv_bs = 1.0f / (float)BS;   // fold batch-mean into gates
        int se[TOPK] = { s0, s1, s2, s3 };
        #pragma unroll
        for (int k = 0; k < TOPK; ++k) {
            int e   = se[k];
            int pos = atomicAdd(&cnt[e], 1);
            tok_list[e * CAP + pos]  = b;
            gate_list[e * CAP + pos] = tv[k] * inv_bs;
        }
    }
}

// --- 5) grouped expert GEMM: bf16 WMMA, gate-weighted column reduction -----
// grid: (token_tile [128], out_tile [4], expert [32]); block: 256 = 8 waves.
__global__ __launch_bounds__(256) void moe_expert_gemm_kernel(
    const unsigned short* __restrict__ xb,   // [BS, DIM] bf16
    const unsigned short* __restrict__ web,  // [NEXP, ODIM, DIM] bf16
    const float* __restrict__ be,            // [NEXP, ODIM]
    const int*   __restrict__ cnt,
    const int*   __restrict__ tok_list,
    const float* __restrict__ gate_list,
    float* __restrict__ out)                 // [ODIM] accumulator
{
    const int e   = blockIdx.z;
    const int mt  = blockIdx.x;
    const int nt  = blockIdx.y;
    const int n_e = cnt[e];
    if (mt * 64 >= n_e) return;              // uniform early exit

#if HAVE_ASYNC_LDS
    __shared__ unsigned short As[2][64 * 32];    // 8 KB  (double buffered)
    __shared__ unsigned short Bs[2][128 * 32];   // 16 KB (double buffered)
#else
    __shared__ unsigned short As1[64 * 32];      // 4 KB
    __shared__ unsigned short Bs1[128 * 32];     // 8 KB
#endif
    __shared__ int   tok_s[64];
    __shared__ float gate_s[64];

    const int t = threadIdx.x;
    if (t < 64) {
        int idx  = mt * 64 + t;
        int good = idx < n_e;
        tok_s[t]  = good ? tok_list[e * CAP + idx] : 0;
        gate_s[t] = good ? gate_list[e * CAP + idx] : 0.0f;  // masks junk rows
    }
    __syncthreads();

    const int wave = t >> 5;
    const int lane = t & 31;
    const int msub = wave >> 1;              // 0..3  (16-row sub-tile)
    const int nq   = wave & 1;               // 0..1  (64-col half -> 4 sub-tiles)
    const int half = (lane < 16) ? 0 : 1;
    const int rowA = msub * 16 + (lane & 15);
    const int aoff = rowA * 32 + half * 8;                      // + {0, 16}
    const int boff = (nq * 64 + (lane & 15)) * 32 + half * 8;   // + {0, 16} + s*512

    const int ldr = t >> 2;                  // 0..63: tile row to load
    const int seg = (t & 3) * 8;             // 8 bf16 (16B) per thread
    const size_t wbase = ((size_t)e * ODIM + (size_t)nt * 128) * DIM;
    const size_t arow  = (size_t)tok_s[ldr] * DIM + seg;
    const size_t brow1 = wbase + (size_t)ldr * DIM + seg;
    const size_t brow2 = wbase + (size_t)(64 + ldr) * DIM + seg;

    v8f acc[4] = {};

#if HAVE_ASYNC_LDS
    // -------- async double-buffered pipeline (ASYNCcnt-tracked DMA) --------
    #define MOE_FILL(bufidx, kk) do {                                          \
        __builtin_amdgcn_global_load_async_to_lds_b128(                        \
            (gvp_b128)(xb + arow + (kk)),                                      \
            (lvp_b128)(&As[bufidx][ldr * 32 + seg]), 0, 0);                    \
        __builtin_amdgcn_global_load_async_to_lds_b128(                        \
            (gvp_b128)(web + brow1 + (kk)),                                    \
            (lvp_b128)(&Bs[bufidx][ldr * 32 + seg]), 0, 0);                    \
        __builtin_amdgcn_global_load_async_to_lds_b128(                        \
            (gvp_b128)(web + brow2 + (kk)),                                    \
            (lvp_b128)(&Bs[bufidx][(64 + ldr) * 32 + seg]), 0, 0);             \
    } while (0)

    MOE_FILL(0, 0);
    MOE_FILL(1, 32);
    #pragma unroll
    for (int i = 0; i < KSTEPS; ++i) {
        const int cur = i & 1;
        if (i == KSTEPS - 1) wait_async_le0();   // last slab: drain everything
        else                 wait_async_le3();   // keep next slab's 3 ops in flight
        __syncthreads();                         // slab i visible to all waves

        FragU a;
        a.u[0] = *(const uint4*)(&As[cur][aoff]);
        a.u[1] = *(const uint4*)(&As[cur][aoff + 16]);
        FragU bf[4];
        #pragma unroll
        for (int s = 0; s < 4; ++s) {
            bf[s].u[0] = *(const uint4*)(&Bs[cur][boff + s * 512]);
            bf[s].u[1] = *(const uint4*)(&Bs[cur][boff + s * 512 + 16]);
        }
        __syncthreads();                         // all waves done reading buf[cur]
        if (i + 2 < KSTEPS) MOE_FILL(cur, (i + 2) * 32);

        #pragma unroll
        for (int s = 0; s < 4; ++s)
            acc[s] = __builtin_amdgcn_wmma_f32_16x16x32_bf16(
                         false, a.v, false, bf[s].v, (short)0, acc[s], false, false);
    }
    #undef MOE_FILL
#else
    // -------- register-staged fallback (compiler software-pipelines) -------
    for (int k0 = 0; k0 < DIM; k0 += 32) {
        uint4 av  = *(const uint4*)(xb  + arow  + k0);
        uint4 bv1 = *(const uint4*)(web + brow1 + k0);
        uint4 bv2 = *(const uint4*)(web + brow2 + k0);
        if (k0 + 32 < DIM) {
            __builtin_prefetch(xb  + arow  + k0 + 32, 0, 1);
            __builtin_prefetch(web + brow1 + k0 + 32, 0, 1);
            __builtin_prefetch(web + brow2 + k0 + 32, 0, 1);
        }
        __syncthreads();
        *(uint4*)(As1 + ldr * 32 + seg)        = av;
        *(uint4*)(Bs1 + ldr * 32 + seg)        = bv1;
        *(uint4*)(Bs1 + (64 + ldr) * 32 + seg) = bv2;
        __syncthreads();

        FragU a;
        a.u[0] = *(const uint4*)(&As1[aoff]);
        a.u[1] = *(const uint4*)(&As1[aoff + 16]);
        FragU bf[4];
        #pragma unroll
        for (int s = 0; s < 4; ++s) {
            bf[s].u[0] = *(const uint4*)(&Bs1[boff + s * 512]);
            bf[s].u[1] = *(const uint4*)(&Bs1[boff + s * 512 + 16]);
        }
        #pragma unroll
        for (int s = 0; s < 4; ++s)
            acc[s] = __builtin_amdgcn_wmma_f32_16x16x32_bf16(
                         false, a.v, false, bf[s].v, (short)0, acc[s], false, false);
    }
#endif

    // epilogue: out[col] += sum_rows gate[row] * sigmoid(acc + bias)
    // C layout: VGPR r -> row r (lane<16) or r+8 (lane>=16); col = lane&15
    const int n0    = nt * 128;
    const int rbase = msub * 16 + half * 8;
    #pragma unroll
    for (int s = 0; s < 4; ++s) {
        int colg   = n0 + nq * 64 + s * 16 + (lane & 15);
        float bias = be[e * ODIM + colg];
        float p = 0.f;
        #pragma unroll
        for (int r = 0; r < 8; ++r)
            p += gate_s[rbase + r] * (1.0f / (1.0f + __expf(-(acc[s][r] + bias))));
        atomicAdd(out + colg, p);
    }
}

// ---------------------------------------------------------------------------
extern "C" void kernel_launch(void* const* d_in, const int* in_sizes, int n_in,
                              void* d_out, int out_size, void* d_ws, size_t ws_size,
                              hipStream_t stream) {
    (void)in_sizes; (void)n_in; (void)out_size; (void)ws_size;
    const float* x  = (const float*)d_in[0];   // [8192, 512]
    const float* Wg = (const float*)d_in[1];   // [32, 512]
    const float* Wn = (const float*)d_in[2];   // [32, 512]
    const float* We = (const float*)d_in[3];   // [32, 512, 512]
    const float* be = (const float*)d_in[4];   // [32, 512]
    float* out = (float*)d_out;                // [512]

    // workspace layout
    char* ws = (char*)d_ws;
    int*   cnt       = (int*)ws;                                   // 128 B
    int*   tok_list  = (int*)(ws + 256);                           // 1 MB
    float* gate_list = (float*)(ws + 256 + (size_t)NEXP * CAP * 4);// 1 MB
    unsigned short* xb  = (unsigned short*)(ws + 256 + (size_t)2 * NEXP * CAP * 4);
    unsigned short* web = xb + (size_t)BS * DIM;                   // 8 MB after xb

    // 1) zero accumulators
    moe_init_kernel<<<1, 512, 0, stream>>>(out, cnt);

    // 2) x -> bf16
    {
        int n4 = BS * DIM / 4;
        cvt_bf16_kernel<<<(n4 + 255) / 256, 256, 0, stream>>>(x, xb, n4);
    }
    // 3) We -> bf16
    {
        int n4 = NEXP * ODIM * DIM / 4;
        cvt_bf16_kernel<<<(n4 + 255) / 256, 256, 0, stream>>>(We, web, n4);
    }
    // 4) routing (f32 fidelity for expert selection)
    moe_route_kernel<<<BS / 8, 256, 0, stream>>>(x, Wg, Wn, cnt, tok_list, gate_list);

    // 5) grouped expert GEMM with WMMA (M=64 x N=128 tiles)
    dim3 grid(CAP / 64, ODIM / 128, NEXP);
    moe_expert_gemm_kernel<<<grid, 256, 0, stream>>>(xb, web, be, cnt,
                                                     tok_list, gate_list, out);
}
